// TransformerBlock_65670049955920
// MI455X (gfx1250) — matrix operand order repaired and checked by reading, not compiled, not run on hardware
//
#include <hip/hip_runtime.h>
#include <hip/hip_bf16.h>
#include <math.h>

// ---------------------------------------------------------------------------
// Transformer block (pre-norm attention + FFN) for MI455X / gfx1250.
// bf16 WMMA (V_WMMA_F32_16X16X32_BF16), async global->LDS tile staging
// (GLOBAL_LOAD_ASYNC_TO_LDS_B128 + ASYNCcnt), double-buffered GEMM pipeline.
// ---------------------------------------------------------------------------

typedef unsigned short u16;
typedef unsigned int   u32;
typedef unsigned long long u64;
typedef __bf16 bf16_t;
typedef bf16_t v16bf __attribute__((ext_vector_type(16)));
typedef float  v8f   __attribute__((ext_vector_type(8)));
typedef u16    v8u   __attribute__((ext_vector_type(8)));
typedef u16    v16u  __attribute__((ext_vector_type(16)));

static constexpr int BATCH = 4;
static constexpr int SEQ   = 2048;
static constexpr int EMB   = 2048;
static constexpr int HEADS = 16;
static constexpr int HDIM  = 128;
static constexpr int FF    = 4 * EMB;          // 8192
static constexpr int ROWS  = BATCH * SEQ;      // 8192

// ---- helpers --------------------------------------------------------------

__device__ __forceinline__ u16 f2bf(float f) {
  u32 u = __builtin_bit_cast(u32, f);
  u32 r = u + 0x7FFFu + ((u >> 16) & 1u);      // round-to-nearest-even
  return (u16)(r >> 16);
}

__device__ __forceinline__ float gelu_tanh(float x) {
  const float c = 0.7978845608028654f;         // sqrt(2/pi)
  return 0.5f * x * (1.0f + tanhf(c * (x + 0.044715f * x * x * x)));
}

__device__ __forceinline__ v8f zero8() {
  v8f z = {0.f, 0.f, 0.f, 0.f, 0.f, 0.f, 0.f, 0.f};
  return z;
}

__device__ __forceinline__ v16bf make_frag(v8u lo, v8u hi) {
  v16u t;
#pragma unroll
  for (int i = 0; i < 8; ++i) { t[i] = lo[i]; t[8 + i] = hi[i]; }
  return __builtin_bit_cast(v16bf, t);
}

// A fragment (16x32 bf16, ISA 7.12.2): lane L holds row M=L&15.
// lanes 0-15: K = kbase+[0..7] and kbase+[16..23]; lanes 16-31: +8.
__device__ __forceinline__ v16bf lds_load_a(const u16* base, int stride,
                                            int mbase, int kbase, int lane) {
  int m  = mbase + (lane & 15);
  int k0 = kbase + ((lane >> 4) << 3);
  const u16* p = base + m * stride + k0;
  v8u lo = *(const v8u*)(p);
  v8u hi = *(const v8u*)(p + 16);
  return make_frag(lo, hi);
}

// B fragment (32x16 bf16): lane L holds column N=L&15;
// lanes 0-15: K = kbase+[0..7] and kbase+[16..23]; lanes 16-31: +8.
// baseT is the B tile stored transposed: baseT[n*stride + k].
__device__ __forceinline__ v16bf lds_load_b(const u16* baseT, int stride,
                                            int kbase, int nbase, int lane) {
  int n  = nbase + (lane & 15);
  int kh = kbase + ((lane >> 4) << 3);
  const u16* p = baseT + n * stride + kh;
  v8u lo = *(const v8u*)(p);
  v8u hi = *(const v8u*)(p + 16);
  return make_frag(lo, hi);
}

__device__ __forceinline__ v8f wmma_bf16(v16bf a, v16bf b, v8f c) {
  return __builtin_amdgcn_wmma_f32_16x16x32_bf16(
      /*neg_a=*/false, a, /*neg_b=*/false, b,
      /*c_mod=*/(short)0, c, /*reuse_a=*/false, /*reuse_b=*/false);
}

// ---- CDNA5 async global->LDS (ASYNCcnt) -----------------------------------
// Per-lane copy of 16 bytes directly into LDS, no VGPR round trip.
// ldsoff = LDS byte offset (low 32 bits of the generic/flat shared address).

__device__ __forceinline__ void async_ld16(u32 ldsoff, const u16* g) {
  u64 ga = (u64)(uintptr_t)g;
  asm volatile("global_load_async_to_lds_b128 %0, %1, off"
               :: "v"(ldsoff), "v"(ga) : "memory");
}
__device__ __forceinline__ void wait_async0() {
  asm volatile("s_wait_asynccnt 0x0" ::: "memory");
}
__device__ __forceinline__ u32 lds_off(const void* p) {
  return (u32)(uintptr_t)p;
}

// ---- fp32 -> bf16 conversion ---------------------------------------------

__global__ void cvt_f32_bf16_kernel(const float* __restrict__ in,
                                    u16* __restrict__ out, int n) {
  int i = blockIdx.x * blockDim.x + threadIdx.x;
  int stride = gridDim.x * blockDim.x;
  for (; i < n; i += stride) out[i] = f2bf(in[i]);
}

// ---- LayerNorm (one block per row of 2048) -> bf16 ------------------------

__global__ void ln_bf16_kernel(const float* __restrict__ x,
                               const float* __restrict__ sc,
                               const float* __restrict__ sh,
                               u16* __restrict__ out, int D) {
  const int row = blockIdx.x, t = threadIdx.x;
  const float* xr = x + (size_t)row * D;
  float s = 0.f, s2 = 0.f;
  for (int i = t; i < D; i += 256) { float v = xr[i]; s += v; s2 += v * v; }
  __shared__ float r1[256], r2[256];
  r1[t] = s; r2[t] = s2; __syncthreads();
  for (int o = 128; o > 0; o >>= 1) {
    if (t < o) { r1[t] += r1[t + o]; r2[t] += r2[t + o]; }
    __syncthreads();
  }
  float mean = r1[0] / (float)D;
  float var  = r2[0] / (float)D - mean * mean;
  float inv  = rsqrtf(var + 1e-5f);
  for (int i = t; i < D; i += 256) {
    float v = (xr[i] - mean) * inv * sc[i] + sh[i];
    out[(size_t)row * D + i] = f2bf(v);
  }
}

// ---- Tiled WMMA GEMM: C[M,N] = act(A[M,K]@B[K,N] + bias) (+resid) ---------
// 128x128 C tile / block (256 thr = 8 waves; wave grid 4x2, 32x64 per wave).
// Double-buffered LDS; A tile staged with async-to-LDS, B tile transposed
// through registers and software-pipelined across the WMMA phase.

template <bool OUT_BF16, bool DO_GELU, bool DO_RES>
__global__ __launch_bounds__(256, 1)
void gemm_bf16_kernel(const u16* __restrict__ A,
                      const u16* __restrict__ B,
                      const float* __restrict__ bias,
                      const float* __restrict__ resid,
                      void* __restrict__ outp,
                      int M, int N, int K) {
  __shared__ __align__(16) u16 At[2][128 * 48];  // [m][k], stride 48
  __shared__ __align__(16) u16 Bt[2][128 * 48];  // transposed [n][k]
  const int t = threadIdx.x, lane = t & 31, w = t >> 5;
  const int nBase = blockIdx.x * 128, mBase = blockIdx.y * 128;
  const int waveM = (w & 3) * 32, waveN = (w >> 2) * 64;

  const u32 atOff[2] = {lds_off(&At[0][0]), lds_off(&At[1][0])};

  // per-thread chunk coordinates (two chunks each for 512-chunk tiles)
  const int ar0 = (t + 0)   >> 2, ak0 = ((t + 0)   & 3) * 8;
  const int ar1 = (t + 256) >> 2, ak1 = ((t + 256) & 3) * 8;
  const int bk0 = (t + 0)   >> 4, bn0 = ((t + 0)   & 15) * 8;
  const int bk1 = (t + 256) >> 4, bn1 = ((t + 256) & 15) * 8;

  v8f acc[2][4];
#pragma unroll
  for (int mi = 0; mi < 2; ++mi)
#pragma unroll
    for (int ni = 0; ni < 4; ++ni) acc[mi][ni] = zero8();

  v8u breg0, breg1;

  // ---- prologue: stage tile 0 ----
  async_ld16(atOff[0] + (ar0 * 48 + ak0) * 2,
             &A[(size_t)(mBase + ar0) * K + ak0]);
  async_ld16(atOff[0] + (ar1 * 48 + ak1) * 2,
             &A[(size_t)(mBase + ar1) * K + ak1]);
  breg0 = *(const v8u*)&B[(size_t)bk0 * N + nBase + bn0];
  breg1 = *(const v8u*)&B[(size_t)bk1 * N + nBase + bn1];
#pragma unroll
  for (int e = 0; e < 8; ++e) {
    Bt[0][(bn0 + e) * 48 + bk0] = breg0[e];
    Bt[0][(bn1 + e) * 48 + bk1] = breg1[e];
  }
  wait_async0();
  __syncthreads();

  int buf = 0;
  for (int k0 = 0; k0 < K; k0 += 32) {
    const int nxt = buf ^ 1;
    const bool hasNext = (k0 + 32) < K;
    if (hasNext) {
      const int k1 = k0 + 32;
      async_ld16(atOff[nxt] + (ar0 * 48 + ak0) * 2,
                 &A[(size_t)(mBase + ar0) * K + k1 + ak0]);
      async_ld16(atOff[nxt] + (ar1 * 48 + ak1) * 2,
                 &A[(size_t)(mBase + ar1) * K + k1 + ak1]);
      breg0 = *(const v8u*)&B[(size_t)(k1 + bk0) * N + nBase + bn0];
      breg1 = *(const v8u*)&B[(size_t)(k1 + bk1) * N + nBase + bn1];
    }

    // ---- compute on current buffer ----
    const u16* Ab = At[buf];
    const u16* Bb = Bt[buf];
    v16bf af0 = lds_load_a(Ab, 48, waveM, 0, lane);
    v16bf af1 = lds_load_a(Ab, 48, waveM + 16, 0, lane);
    v16bf bf0 = lds_load_b(Bb, 48, 0, waveN + 0,  lane);
    v16bf bf1 = lds_load_b(Bb, 48, 0, waveN + 16, lane);
    v16bf bf2 = lds_load_b(Bb, 48, 0, waveN + 32, lane);
    v16bf bf3 = lds_load_b(Bb, 48, 0, waveN + 48, lane);
    acc[0][0] = wmma_bf16(af0, bf0, acc[0][0]);
    acc[1][0] = wmma_bf16(af1, bf0, acc[1][0]);
    acc[0][1] = wmma_bf16(af0, bf1, acc[0][1]);
    acc[1][1] = wmma_bf16(af1, bf1, acc[1][1]);
    acc[0][2] = wmma_bf16(af0, bf2, acc[0][2]);
    acc[1][2] = wmma_bf16(af1, bf2, acc[1][2]);
    acc[0][3] = wmma_bf16(af0, bf3, acc[0][3]);
    acc[1][3] = wmma_bf16(af1, bf3, acc[1][3]);

    if (hasNext) {
#pragma unroll
      for (int e = 0; e < 8; ++e) {
        Bt[nxt][(bn0 + e) * 48 + bk0] = breg0[e];
        Bt[nxt][(bn1 + e) * 48 + bk1] = breg1[e];
      }
      wait_async0();
    }
    __syncthreads();
    buf = nxt;
  }

  // Epilogue. C layout: row = r + 8*(lane>=16), col = lane&15 (per 16x16 frag)
  const int rsel = (lane >> 4) << 3;
  const int csel = lane & 15;
#pragma unroll
  for (int mi = 0; mi < 2; ++mi) {
#pragma unroll
    for (int ni = 0; ni < 4; ++ni) {
      int col = nBase + waveN + ni * 16 + csel;
      float bv = bias ? bias[col] : 0.f;
#pragma unroll
      for (int r = 0; r < 8; ++r) {
        int row = mBase + waveM + mi * 16 + r + rsel;
        float v = acc[mi][ni][r] + bv;
        if (DO_GELU) v = gelu_tanh(v);
        if (DO_RES)  v += resid[(size_t)row * N + col];
        if (OUT_BF16) ((u16*)outp)[(size_t)row * N + col] = f2bf(v);
        else          ((float*)outp)[(size_t)row * N + col] = v;
      }
    }
  }
}

// ---- Flash attention: one block per (q-tile 128, head, batch) -------------
// q/k/v are bf16 [ROWS][EMB] with head h occupying columns h*128..h*128+127.

__global__ __launch_bounds__(256, 1)
void flash_attn_kernel(const u16* __restrict__ qb,
                       const u16* __restrict__ kb,
                       const u16* __restrict__ vb,
                       u16* __restrict__ ctx) {
  __shared__ __align__(16) u16 Qs[128 * 136];  // [qrow][d]
  __shared__ __align__(16) u16 Ks[128 * 136];  // [key][d], reused as P
  __shared__ __align__(16) u16 Vt[128 * 136];  // V transposed: [d][key]
  const int t = threadIdx.x, lane = t & 31, w = t >> 5;
  const int qt = blockIdx.x, head = blockIdx.y, bat = blockIdx.z;
  const size_t rowOff = (size_t)bat * SEQ;
  const int colOff = head * HDIM;
  const int rsel = (lane >> 4) << 3;
  const int csel = lane & 15;
  const float SCALE = 0.08838834764831845f;    // 1/sqrt(128)
  const u32 qsOff = lds_off(Qs), ksOff = lds_off(Ks);

  // stage Q tile (128x128 bf16) via async copies
#pragma unroll
  for (int it = 0; it < 8; ++it) {
    int c = t + it * 256;
    int r = c >> 4, d0 = (c & 15) * 8;
    async_ld16(qsOff + (r * 136 + d0) * 2,
               &qb[(rowOff + qt * 128 + r) * EMB + colOff + d0]);
  }

  float rmax[8], rsum[8];
  v8f accO[8];
#pragma unroll
  for (int r = 0; r < 8; ++r) { rmax[r] = -1e30f; rsum[r] = 0.f; }
#pragma unroll
  for (int f = 0; f < 8; ++f) accO[f] = zero8();

  for (int kt = 0; kt <= qt; ++kt) {
    // stage K tile async; V tile transposed through registers
#pragma unroll
    for (int it = 0; it < 8; ++it) {
      int c = t + it * 256;
      int r = c >> 4, d0 = (c & 15) * 8;
      async_ld16(ksOff + (r * 136 + d0) * 2,
                 &kb[(rowOff + kt * 128 + r) * EMB + colOff + d0]);
      v8u vv = *(const v8u*)&vb[(rowOff + kt * 128 + r) * EMB + colOff + d0];
#pragma unroll
      for (int e = 0; e < 8; ++e) Vt[(d0 + e) * 136 + r] = vv[e];
    }
    wait_async0();
    __syncthreads();

    // S = Q @ K^T for this wave's 16 query rows x 128 keys
    v8f accS[8];
#pragma unroll
    for (int f = 0; f < 8; ++f) accS[f] = zero8();
#pragma unroll
    for (int kc = 0; kc < 4; ++kc) {
      v16bf aq = lds_load_a(Qs, 136, w * 16, kc * 32, lane);
#pragma unroll
      for (int fp = 0; fp < 4; ++fp) {
        v16bf b0 = lds_load_b(Ks, 136, kc * 32, (2 * fp) * 16, lane);
        v16bf b1 = lds_load_b(Ks, 136, kc * 32, (2 * fp + 1) * 16, lane);
        accS[2 * fp]     = wmma_bf16(aq, b0, accS[2 * fp]);
        accS[2 * fp + 1] = wmma_bf16(aq, b1, accS[2 * fp + 1]);
      }
    }
    __syncthreads();   // all waves done reading Ks before reuse as P

    // online softmax (rows = w*16 + r + rsel, cols = kt*128 + f*16 + csel)
    float nm[8];
#pragma unroll
    for (int r = 0; r < 8; ++r) nm[r] = rmax[r];
#pragma unroll
    for (int f = 0; f < 8; ++f) {
      int col = kt * 128 + f * 16 + csel;
#pragma unroll
      for (int r = 0; r < 8; ++r) {
        int row = qt * 128 + w * 16 + r + rsel;
        float s = accS[f][r] * SCALE;
        if (col > row) s = -1e30f;             // causal mask
        accS[f][r] = s;
        nm[r] = fmaxf(nm[r], s);
      }
    }
#pragma unroll
    for (int m = 1; m <= 8; m <<= 1)
#pragma unroll
      for (int r = 0; r < 8; ++r)
        nm[r] = fmaxf(nm[r], __shfl_xor(nm[r], m, 32));

    float corr[8], psum[8];
#pragma unroll
    for (int r = 0; r < 8; ++r) {
      corr[r] = __expf(rmax[r] - nm[r]);
      rmax[r] = nm[r];
      psum[r] = 0.f;
    }
#pragma unroll
    for (int f = 0; f < 8; ++f)
#pragma unroll
      for (int r = 0; r < 8; ++r) {
        float p = __expf(accS[f][r] - nm[r]);
        accS[f][r] = p;
        psum[r] += p;
      }
#pragma unroll
    for (int m = 1; m <= 8; m <<= 1)
#pragma unroll
      for (int r = 0; r < 8; ++r) psum[r] += __shfl_xor(psum[r], m, 32);
#pragma unroll
    for (int r = 0; r < 8; ++r) rsum[r] = rsum[r] * corr[r] + psum[r];
#pragma unroll
    for (int f = 0; f < 8; ++f)
#pragma unroll
      for (int r = 0; r < 8; ++r) accO[f][r] *= corr[r];

    // write P (bf16) into Ks region: P[qrow_local][key]
#pragma unroll
    for (int f = 0; f < 8; ++f)
#pragma unroll
      for (int r = 0; r < 8; ++r)
        Ks[(w * 16 + r + rsel) * 136 + f * 16 + csel] = f2bf(accS[f][r]);
    __syncthreads();

    // O += P @ V
#pragma unroll
    for (int kc = 0; kc < 4; ++kc) {
      v16bf ap = lds_load_a(Ks, 136, w * 16, kc * 32, lane);
#pragma unroll
      for (int fp = 0; fp < 4; ++fp) {
        v16bf b0 = lds_load_b(Vt, 136, kc * 32, (2 * fp) * 16, lane);
        v16bf b1 = lds_load_b(Vt, 136, kc * 32, (2 * fp + 1) * 16, lane);
        accO[2 * fp]     = wmma_bf16(ap, b0, accO[2 * fp]);
        accO[2 * fp + 1] = wmma_bf16(ap, b1, accO[2 * fp + 1]);
      }
    }
    __syncthreads();   // before next kt overwrites Ks / Vt
  }

  // normalize and store ctx (bf16, [row][EMB])
#pragma unroll
  for (int f = 0; f < 8; ++f) {
    int d = colOff + f * 16 + csel;
#pragma unroll
    for (int r = 0; r < 8; ++r) {
      int row = qt * 128 + w * 16 + r + rsel;
      float o = accO[f][r] / rsum[r];
      ctx[(rowOff + row) * EMB + d] = f2bf(o);
    }
  }
}

// ---------------------------------------------------------------------------
// Host-side orchestration
// ---------------------------------------------------------------------------

extern "C" void kernel_launch(void* const* d_in, const int* in_sizes, int n_in,
                              void* d_out, int out_size, void* d_ws,
                              size_t ws_size, hipStream_t stream) {
  (void)in_sizes; (void)n_in; (void)out_size; (void)ws_size;

  const float* x    = (const float*)d_in[0];
  const float* wq   = (const float*)d_in[1];
  const float* wk   = (const float*)d_in[2];
  const float* wv   = (const float*)d_in[3];
  const float* wo   = (const float*)d_in[4];
  const float* bo   = (const float*)d_in[5];
  const float* w1   = (const float*)d_in[6];
  const float* b1   = (const float*)d_in[7];
  const float* w2   = (const float*)d_in[8];
  const float* b2   = (const float*)d_in[9];
  const float* ln1s = (const float*)d_in[10];
  const float* ln1b = (const float*)d_in[11];
  const float* ln2s = (const float*)d_in[12];
  const float* ln2b = (const float*)d_in[13];
  float* out = (float*)d_out;

  char* ws = (char*)d_ws;
  const size_t MB = 1024ull * 1024ull;
  u16*   h    = (u16*)(ws + 0);         // 32 MB  bf16 [ROWS][EMB]
  u16*   wqb  = (u16*)(ws + 32  * MB);  // 8 MB each
  u16*   wkb  = (u16*)(ws + 40  * MB);
  u16*   wvb  = (u16*)(ws + 48  * MB);
  u16*   wob  = (u16*)(ws + 56  * MB);
  u16*   w1b  = (u16*)(ws + 64  * MB);  // 32 MB
  u16*   w2b  = (u16*)(ws + 96  * MB);  // 32 MB
  float* x1   = (float*)(ws + 128 * MB);// 64 MB  fp32 [ROWS][EMB]
  u16*   qbuf = (u16*)(ws + 192 * MB);  // 32 MB
  u16*   kbuf = (u16*)(ws + 224 * MB);  // 32 MB
  u16*   vbuf = (u16*)(ws + 256 * MB);  // 32 MB
  u16*   ctx  = (u16*)(ws + 288 * MB);  // 32 MB
  u16*   g    = (u16*)(ws + 192 * MB);  // 128 MB (reuses q/k/v/ctx region)

  // 1) weights fp32 -> bf16
  cvt_f32_bf16_kernel<<<1024, 256, 0, stream>>>(wq, wqb, EMB * EMB);
  cvt_f32_bf16_kernel<<<1024, 256, 0, stream>>>(wk, wkb, EMB * EMB);
  cvt_f32_bf16_kernel<<<1024, 256, 0, stream>>>(wv, wvb, EMB * EMB);
  cvt_f32_bf16_kernel<<<1024, 256, 0, stream>>>(wo, wob, EMB * EMB);
  cvt_f32_bf16_kernel<<<1024, 256, 0, stream>>>(w1, w1b, EMB * FF);
  cvt_f32_bf16_kernel<<<1024, 256, 0, stream>>>(w2, w2b, FF * EMB);

  // 2) h = LN1(x)
  ln_bf16_kernel<<<ROWS, 256, 0, stream>>>(x, ln1s, ln1b, h, EMB);

  // 3) q/k/v projections (bf16 out)
  dim3 gProj(EMB / 128, ROWS / 128);
  gemm_bf16_kernel<true, false, false><<<gProj, 256, 0, stream>>>(
      h, wqb, nullptr, nullptr, qbuf, ROWS, EMB, EMB);
  gemm_bf16_kernel<true, false, false><<<gProj, 256, 0, stream>>>(
      h, wkb, nullptr, nullptr, kbuf, ROWS, EMB, EMB);
  gemm_bf16_kernel<true, false, false><<<gProj, 256, 0, stream>>>(
      h, wvb, nullptr, nullptr, vbuf, ROWS, EMB, EMB);

  // 4) causal flash attention -> ctx (bf16)
  flash_attn_kernel<<<dim3(SEQ / 128, HEADS, BATCH), 256, 0, stream>>>(
      qbuf, kbuf, vbuf, ctx);

  // 5) x1 = x + ctx @ wo + bo   (fp32)
  gemm_bf16_kernel<false, false, true><<<gProj, 256, 0, stream>>>(
      ctx, wob, bo, x, x1, ROWS, EMB, EMB);

  // 6) h = LN2(x1)
  ln_bf16_kernel<<<ROWS, 256, 0, stream>>>(x1, ln2s, ln2b, h, EMB);

  // 7) g = gelu(h @ w1 + b1)    (bf16)
  gemm_bf16_kernel<true, true, false><<<dim3(FF / 128, ROWS / 128), 256, 0,
                                        stream>>>(
      h, w1b, b1, nullptr, g, ROWS, FF, EMB);

  // 8) out = x1 + g @ w2 + b2   (fp32, final)
  gemm_bf16_kernel<false, false, true><<<gProj, 256, 0, stream>>>(
      g, w2b, b2, x1, out, ROWS, EMB, FF);
}
